// Mamba1_11957188952408
// MI455X (gfx1250) — compile-verified, matrix operand order checked by reading, not run on hardware
//
#include <hip/hip_runtime.h>
#include <cstdint>
#include <cstddef>

// ---------------- problem constants ----------------
#define BSZ      2
#define SEQ      4096
#define DMODEL   1024
#define DINNER   2048
#define DSTATE   16
#define NCHUNK   16
#define CHUNK    256   // NCHUNK*CHUNK == SEQ

// ---------------- WMMA types ----------------
typedef __attribute__((ext_vector_type(16))) __bf16 bf16x16;
typedef __attribute__((ext_vector_type(8)))  __bf16 bf16x8;
typedef __attribute__((ext_vector_type(8)))  float  f32x8;

__device__ __forceinline__ float silu(float x) {
  return x / (1.f + __expf(-x));
}

// ============================================================================
// bf16 WMMA GEMM: C[M,N] = A[M,K] * B[N,K]^T (+bias, +activation)
// Block tile 128x128, BK=32, 256 threads (8 waves).
// Wave tile 32x64: 2 A-frags x 4 B-frags = 8 v_wmma per K-step.
// Double-buffered LDS (2 x 20KB), global loads for step k+1 issued before the
// WMMAs of step k. One barrier per K-step. f32->bf16 via hardware cvt
// (__builtin_convertvector -> packed cvt), not manual bit twiddling.
// ============================================================================
#define BM 128
#define BN 128
#define BK 32
#define LPITCH 40   // bf16 elems per LDS row (80B = 16B multiple)

template <int ACT, bool HAS_BIAS>
__global__ __launch_bounds__(256) void wmma_gemm_nt(
    const float* __restrict__ A, int lda,
    const float* __restrict__ Bm, int ldb,
    float* __restrict__ C, int ldc,
    int K,
    const float* __restrict__ bias1,
    const float* __restrict__ bias2)
{
  __shared__ unsigned short Alds[2][BM * LPITCH];
  __shared__ unsigned short Blds[2][BN * LPITCH];

  const int tid  = threadIdx.x;
  const int lane = tid & 31;
  const int wave = tid >> 5;
  const int g    = lane >> 4;   // half-wave group
  const int ln   = lane & 15;

  const int m_tile = blockIdx.y * BM;
  const int n_tile = blockIdx.x * BN;
  const int wm = (wave & 3) * 32;    // wave row offset (4 wave-rows)
  const int wn = (wave >> 2) * 64;   // wave col offset (2 wave-cols)

  f32x8 acc[2][4];
#pragma unroll
  for (int i = 0; i < 2; ++i)
#pragma unroll
    for (int j = 0; j < 4; ++j) acc[i][j] = (f32x8){};

  // staging assignment: each thread moves 16 f32 of A and 16 f32 of B per step
  const int srow = tid >> 1;         // 0..127
  const int scol = (tid & 1) * 16;   // 0 or 16

  const float* aptr = A + (size_t)(m_tile + srow) * lda + scol;
  const float* bptr = Bm + (size_t)(n_tile + srow) * ldb + scol;

  f32x8 areg[2], breg[2];

  // ---- prologue: global load K-step 0 ----
#pragma unroll
  for (int h = 0; h < 2; ++h) {
    areg[h] = *(const f32x8*)(aptr + h * 8);
    breg[h] = *(const f32x8*)(bptr + h * 8);
  }
#pragma unroll
  for (int h = 0; h < 2; ++h) {
    *(bf16x8*)&Alds[0][srow * LPITCH + scol + h * 8] =
        __builtin_convertvector(areg[h], bf16x8);
    *(bf16x8*)&Blds[0][srow * LPITCH + scol + h * 8] =
        __builtin_convertvector(breg[h], bf16x8);
  }
  __syncthreads();

  union Frag { bf16x16 v; bf16x8 h[2]; };

  int cur = 0;
  for (int kt = 0; kt < K; kt += BK) {
    const bool have_next = (kt + BK) < K;
    // ---- issue global loads for next K-step ----
    if (have_next) {
#pragma unroll
      for (int h = 0; h < 2; ++h) {
        areg[h] = *(const f32x8*)(aptr + (kt + BK) + h * 8);
        breg[h] = *(const f32x8*)(bptr + (kt + BK) + h * 8);
      }
    }

    // ---- fragments from current buffer + 8 WMMAs ----
    Frag af[2], bf[4];
#pragma unroll
    for (int fm = 0; fm < 2; ++fm) {
      const unsigned short* p = &Alds[cur][(wm + fm * 16 + ln) * LPITCH];
      af[fm].h[0] = *(const bf16x8*)(p + g * 8);
      af[fm].h[1] = *(const bf16x8*)(p + 16 + g * 8);
    }
#pragma unroll
    for (int fn = 0; fn < 4; ++fn) {
      const unsigned short* p = &Blds[cur][(wn + fn * 16 + ln) * LPITCH];
      bf[fn].h[0] = *(const bf16x8*)(p + g * 8);
      bf[fn].h[1] = *(const bf16x8*)(p + 16 + g * 8);
    }
#pragma unroll
    for (int fm = 0; fm < 2; ++fm)
#pragma unroll
      for (int fn = 0; fn < 4; ++fn)
        acc[fm][fn] = __builtin_amdgcn_wmma_f32_16x16x32_bf16(
            false, af[fm].v, false, bf[fn].v, (short)0, acc[fm][fn],
            false, false);

    // ---- convert + store next tile into other buffer ----
    if (have_next) {
      const int nxt = cur ^ 1;
#pragma unroll
      for (int h = 0; h < 2; ++h) {
        *(bf16x8*)&Alds[nxt][srow * LPITCH + scol + h * 8] =
            __builtin_convertvector(areg[h], bf16x8);
        *(bf16x8*)&Blds[nxt][srow * LPITCH + scol + h * 8] =
            __builtin_convertvector(breg[h], bf16x8);
      }
    }
    __syncthreads();
    cur ^= 1;
  }

  // ---- epilogue: C/D layout: lane n = l&15, VGPR r -> row r + 8*g ----
#pragma unroll
  for (int fn = 0; fn < 4; ++fn) {
    const int col = n_tile + wn + fn * 16 + ln;
    float badd = 0.f;
    if (HAS_BIAS) {
      badd = bias1[col] + bias2[col];
    }
#pragma unroll
    for (int fm = 0; fm < 2; ++fm) {
      const int row0 = m_tile + wm + fm * 16 + g * 8;
#pragma unroll
      for (int r = 0; r < 8; ++r) {
        float v = acc[fm][fn][r] + badd;
        if (ACT == 1) v = (v > 20.f) ? v : log1pf(__expf(v));  // softplus
        C[(size_t)(row0 + r) * ldc + col] = v;
      }
    }
  }
}

// ---------------- depthwise causal conv (K=4) + SiLU ----------------
// xz: (B*L, 2*DINNER); x half = cols [0,DINNER). u out: (B*L, DINNER)
__global__ __launch_bounds__(256) void conv_silu_kernel(
    const float* __restrict__ xz, const float* __restrict__ cw,
    const float* __restrict__ cb, float* __restrict__ u)
{
  const int idx = blockIdx.x * 256 + threadIdx.x;  // B*L*DINNER
  const int d  = idx & (DINNER - 1);
  const int bl = idx >> 11;
  const int l  = bl & (SEQ - 1);
  float accv = cb[d];
  const float* w = cw + d * 4;
#pragma unroll
  for (int k = 0; k < 4; ++k) {
    const int ll = l - 3 + k;
    if (ll >= 0)
      accv += w[k] * xz[(size_t)(bl - 3 + k) * (2 * DINNER) + d];
  }
  u[idx] = silu(accv);
}

// ---------------- selective scan, 3-pass chunked ----------------
// pass1: per (b,chunk,d) local scan from zero -> final state S[16] and sum(dt)
__global__ __launch_bounds__(256) void scan_pass1(
    const float* __restrict__ dt, const float* __restrict__ u,
    const float* __restrict__ A_log, const float* __restrict__ Bp,
    float* __restrict__ Sout, float* __restrict__ dtsum)
{
  const int idx = blockIdx.x * 256 + threadIdx.x;  // B*NCHUNK*DINNER
  const int d = idx & (DINNER - 1);
  const int c = (idx >> 11) & (NCHUNK - 1);
  const int b = idx >> 15;
  float Av[DSTATE], Bv[DSTATE], s[DSTATE];
#pragma unroll
  for (int n = 0; n < DSTATE; ++n) {
    Av[n] = -__expf(A_log[d * DSTATE + n]);
    Bv[n] = Bp[d * DSTATE + n];
    s[n] = 0.f;
  }
  size_t off = ((size_t)b * SEQ + (size_t)c * CHUNK) * DINNER + d;
  float ds = 0.f;
  for (int t = 0; t < CHUNK; ++t) {
    const float dtv = dt[off];
    const float uv  = u[off];
    off += DINNER;
    ds += dtv;
    const float dtu = dtv * uv;
#pragma unroll
    for (int n = 0; n < DSTATE; ++n)
      s[n] = __expf(dtv * Av[n]) * s[n] + dtu * Bv[n];
  }
  const size_t o = (size_t)idx * DSTATE;
#pragma unroll
  for (int n = 0; n < DSTATE; ++n) Sout[o + n] = s[n];
  dtsum[idx] = ds;
}

// pass2: per (b,d) combine chunk summaries sequentially -> carry-in per chunk
// chunk transition product: prod_t exp(dt_t*A[n]) == exp(A[n] * sum_t dt_t)
__global__ __launch_bounds__(256) void scan_pass2(
    const float* __restrict__ A_log, const float* __restrict__ Sin,
    const float* __restrict__ dtsum, float* __restrict__ carry_in)
{
  const int idx = blockIdx.x * 256 + threadIdx.x;  // B*DINNER
  if (idx >= BSZ * DINNER) return;
  const int d = idx & (DINNER - 1);
  const int b = idx >> 11;
  float Av[DSTATE], carry[DSTATE];
#pragma unroll
  for (int n = 0; n < DSTATE; ++n) {
    Av[n] = -__expf(A_log[d * DSTATE + n]);
    carry[n] = 0.f;
  }
  for (int c = 0; c < NCHUNK; ++c) {
    const size_t cidx = ((size_t)b * NCHUNK + c) * DINNER + d;
    const size_t o = cidx * DSTATE;
    const float ds = dtsum[cidx];
#pragma unroll
    for (int n = 0; n < DSTATE; ++n) {
      carry_in[o + n] = carry[n];
      carry[n] = __expf(Av[n] * ds) * carry[n] + Sin[o + n];
    }
  }
}

// pass3: re-scan each chunk from its carry-in; fuse +D*u and *silu(gate);
// write gated y into the (dead) x-half of xz.
__global__ __launch_bounds__(256) void scan_pass3(
    const float* __restrict__ dt, const float* __restrict__ u,
    float* __restrict__ xz,
    const float* __restrict__ A_log, const float* __restrict__ Bp,
    const float* __restrict__ Cp, const float* __restrict__ Dp,
    const float* __restrict__ carry_in)
{
  const int idx = blockIdx.x * 256 + threadIdx.x;  // B*NCHUNK*DINNER
  const int d = idx & (DINNER - 1);
  const int c = (idx >> 11) & (NCHUNK - 1);
  const int b = idx >> 15;
  float Av[DSTATE], Bv[DSTATE], Cv[DSTATE], s[DSTATE];
#pragma unroll
  for (int n = 0; n < DSTATE; ++n) {
    Av[n] = -__expf(A_log[d * DSTATE + n]);
    Bv[n] = Bp[d * DSTATE + n];
    Cv[n] = Cp[d * DSTATE + n];
    s[n]  = carry_in[(size_t)idx * DSTATE + n];
  }
  const float Dv = Dp[d];
  const size_t base = (size_t)b * SEQ + (size_t)c * CHUNK;
  for (int t = 0; t < CHUNK; ++t) {
    const size_t ro = (base + t) * DINNER + d;
    const float dtv = dt[ro];
    const float uv  = u[ro];
    const float gv  = xz[(base + t) * (2 * DINNER) + DINNER + d];
    const float dtu = dtv * uv;
    float y = 0.f;
#pragma unroll
    for (int n = 0; n < DSTATE; ++n) {
      s[n] = __expf(dtv * Av[n]) * s[n] + dtu * Bv[n];
      y += s[n] * Cv[n];
    }
    y = (y + Dv * uv) * silu(gv);
    xz[(base + t) * (2 * DINNER) + d] = y;  // reuse x-half as gated-y buffer
  }
}

// ---------------- launcher ----------------
extern "C" void kernel_launch(void* const* d_in, const int* in_sizes, int n_in,
                              void* d_out, int out_size, void* d_ws, size_t ws_size,
                              hipStream_t stream) {
  (void)in_sizes; (void)n_in; (void)out_size; (void)ws_size;
  const float* hs      = (const float*)d_in[0];
  const float* W_in    = (const float*)d_in[1];
  const float* W_out   = (const float*)d_in[2];
  const float* W_dt    = (const float*)d_in[3];
  const float* b_dt    = (const float*)d_in[4];
  const float* conv_w  = (const float*)d_in[5];
  const float* conv_b  = (const float*)d_in[6];
  const float* A_log   = (const float*)d_in[7];
  const float* B_par   = (const float*)d_in[8];
  const float* C_par   = (const float*)d_in[9];
  const float* D_par   = (const float*)d_in[10];
  const float* dt_bias = (const float*)d_in[11];

  const size_t BL = (size_t)BSZ * SEQ;  // 8192
  char* ws = (char*)d_ws;
  float* xz = (float*)ws;  ws += BL * (2 * DINNER) * sizeof(float);           // 134 MB
  float* u  = (float*)ws;  ws += BL * DINNER * sizeof(float);                 //  67 MB
  float* dt = (float*)ws;  ws += BL * DINNER * sizeof(float);                 //  67 MB
  float* S  = (float*)ws;  ws += (size_t)BSZ * NCHUNK * DINNER * DSTATE * sizeof(float);
  float* dts= (float*)ws;  ws += (size_t)BSZ * NCHUNK * DINNER * sizeof(float);
  float* ci = (float*)ws;  ws += (size_t)BSZ * NCHUNK * DINNER * DSTATE * sizeof(float);

  const dim3 blk(256);

  // 1) xz = hs @ W_in^T     (8192 x 1024) x (4096 x 1024)^T
  wmma_gemm_nt<0, false><<<dim3((2 * DINNER) / BN, BL / BM), blk, 0, stream>>>(
      hs, DMODEL, W_in, DMODEL, xz, 2 * DINNER, DMODEL, nullptr, nullptr);

  // 2) u = silu(causal_dwconv(x) + conv_b)
  conv_silu_kernel<<<dim3((BL * DINNER) / 256), blk, 0, stream>>>(
      xz, conv_w, conv_b, u);

  // 3) dt = softplus(u @ W_dt^T + b_dt + dt_bias)
  wmma_gemm_nt<1, true><<<dim3(DINNER / BN, BL / BM), blk, 0, stream>>>(
      u, DINNER, W_dt, DINNER, dt, DINNER, DINNER, b_dt, dt_bias);

  // 4) chunked selective scan + fused skip/gate -> gated y into xz x-half
  scan_pass1<<<dim3((BSZ * NCHUNK * DINNER) / 256), blk, 0, stream>>>(
      dt, u, A_log, B_par, S, dts);
  scan_pass2<<<dim3((BSZ * DINNER + 255) / 256), blk, 0, stream>>>(
      A_log, S, dts, ci);
  scan_pass3<<<dim3((BSZ * NCHUNK * DINNER) / 256), blk, 0, stream>>>(
      dt, u, xz, A_log, B_par, C_par, D_par, ci);

  // 5) out = ygated @ W_out^T   (A is pitched: row stride 4096, cols [0,2048))
  wmma_gemm_nt<0, false><<<dim3(DMODEL / BN, BL / BM), blk, 0, stream>>>(
      xz, 2 * DINNER, W_out, DINNER, (float*)d_out, DMODEL, DINNER,
      nullptr, nullptr);
}
